// RiskAttention_73735998538375
// MI455X (gfx1250) — matrix operand order repaired and checked by reading, not compiled, not run on hardware
//
#include <hip/hip_runtime.h>

// ---------------------------------------------------------------------------
// Risk-attention pipeline for MI455X (gfx1250, wave32, WMMA).
// GEMMs: v_wmma_f32_16x16x32_bf16, double-buffered LDS tiles filled with
// CDNA5 async global->LDS copies (global_load_async_to_lds_b128 / ASYNCcnt).
// All B operands stored [N,K] (weights pre-transposed once; V written
// transposed by its producing GEMM) so staging is a rectangular copy and
// every WMMA operand fetch is two contiguous 16B LDS reads per lane.
// ---------------------------------------------------------------------------

typedef __attribute__((ext_vector_type(16))) __bf16 v16bf;
typedef __attribute__((ext_vector_type(8)))  __bf16 v8bf;
typedef __attribute__((ext_vector_type(8)))  float  v8f;

#define BM 128        // block tile M (8 waves x 16 rows)
#define BN 128        // block tile N (8 wmma tiles)
#define BK 32         // K step = one wmma K
#define LDS_STRIDE 40 // padded row stride (elements): 80B = 16B-aligned,
                      // 20-bank lane step -> conflict-free operand reads
#define TILE_ELEMS (128 * LDS_STRIDE)

__device__ __forceinline__ v8f zero8() {
  v8f z = {0.f, 0.f, 0.f, 0.f, 0.f, 0.f, 0.f, 0.f};
  return z;
}

// CDNA5 async copy: 16 bytes global -> LDS, tracked by ASYNCcnt.
__device__ __forceinline__ void async_copy16(unsigned lds_off, const __bf16* g) {
  asm volatile("global_load_async_to_lds_b128 %0, %1, off"
               :: "v"(lds_off), "v"((unsigned long long)(size_t)g)
               : "memory");
}
__device__ __forceinline__ void wait_async0() {
  asm volatile("s_wait_asynccnt 0" ::: "memory");
}

// Stage one BMxBK (or BNxBK) tile: 512 chunks of 16B, 2 per thread per matrix.
__device__ __forceinline__ void stage_async(const __bf16* __restrict__ gA, int Ka,
                                            int m0,
                                            const __bf16* __restrict__ gB, int Kb,
                                            int n0, int k0,
                                            __bf16* sA, __bf16* sB, int t) {
#pragma unroll
  for (int i = 0; i < 2; ++i) {
    int c = t + i * 256;
    int row = c >> 2;          // 4 chunks of 8 elements per row
    int cp = (c & 3) * 8;      // K offset in tile: 0,8,16,24
    unsigned la = (unsigned)(size_t)&sA[row * LDS_STRIDE + cp];
    async_copy16(la, gA + (long)(m0 + row) * Ka + (k0 + cp));
    unsigned lb = (unsigned)(size_t)&sB[row * LDS_STRIDE + cp];
    async_copy16(lb, gB + (long)(n0 + row) * Kb + (k0 + cp));
  }
}

union bf16x16 {
  v16bf v;
  v8bf h[2];
};

// ---------------------------------------------------------------------------
// C[bz][M,N] = alpha * A[bz][M,K] @ B[bz]^T + bias,  B stored [N,K] row-major.
// Outputs: Cf (fp32 row-major), Ch (bf16 row-major), ChT (bf16, per-batch
// transposed: idx = (row/tS)*tBatch + col*tS + row%tS). M%128==N%128==K%32==0.
// ---------------------------------------------------------------------------
__global__ __launch_bounds__(256) void gemm_bf16_kernel(
    const __bf16* __restrict__ A, const __bf16* __restrict__ Bt,
    const float* __restrict__ bias, const float* __restrict__ alphaPtr,
    float* __restrict__ Cf, __bf16* __restrict__ Ch, __bf16* __restrict__ ChT,
    int M, int N, int K, int tS,
    long aBatch, long bBatch, long cBatch, long tBatch) {
  __shared__ __align__(16) __bf16 sA[2][TILE_ELEMS];
  __shared__ __align__(16) __bf16 sB[2][TILE_ELEMS];

  const int t = threadIdx.x;
  const int lane = t & 31;
  const int wave = t >> 5;
  const int bz = blockIdx.z;
  const int m0 = blockIdx.y * BM;
  const int n0 = blockIdx.x * BN;

  A += (long)bz * aBatch;
  Bt += (long)bz * bBatch;

  v8f acc[BN / 16];
#pragma unroll
  for (int i = 0; i < BN / 16; ++i) acc[i] = zero8();

  // Per-lane operand bases (row-major padded LDS tiles).
  // A operand (16x32): lane<16 -> row=lane, K={0..7,16..23};
  //                    lane>=16 -> row=lane-16, K={8..15,24..31}.
  const int arow = wave * 16 + (lane & 15);
  const int akb = (lane < 16) ? 0 : 8;
  // B operand (32x16): lane<16 -> col=lane, K={0..7,16..23}; lane>=16 -> col, K={8..15,24..31}.
  const int bcol = lane & 15;
  const int bkb = (lane < 16) ? 0 : 8;

  stage_async(A, K, m0, Bt, K, n0, 0, sA[0], sB[0], t);
  wait_async0();
  __syncthreads();

  int cur = 0;
  for (int k0 = 0; k0 < K; k0 += BK) {
    if (k0 + BK < K)  // async-fill the other buffer while we compute
      stage_async(A, K, m0, Bt, K, n0, k0 + BK, sA[cur ^ 1], sB[cur ^ 1], t);

    const __bf16* pa = &sA[cur][arow * LDS_STRIDE];
    bf16x16 a;
    a.h[0] = *(const v8bf*)(pa + akb);        // K run 1 -> VGPRs 0-3
    a.h[1] = *(const v8bf*)(pa + 16 + akb);   // K run 2 -> VGPRs 4-7
#pragma unroll
    for (int nt = 0; nt < BN / 16; ++nt) {
      const __bf16* pb = &sB[cur][(nt * 16 + bcol) * LDS_STRIDE + bkb];
      bf16x16 b;
      b.h[0] = *(const v8bf*)(pb);
      b.h[1] = *(const v8bf*)(pb + 16);
      acc[nt] = __builtin_amdgcn_wmma_f32_16x16x32_bf16(
          false, a.v, false, b.v, (short)0, acc[nt], false, false);
    }
    wait_async0();
    __syncthreads();
    cur ^= 1;
  }

  // Epilogue. C layout: VGPR i -> M = i (lanes 0-15) or 8+i (lanes 16-31).
  const float alpha = alphaPtr ? alphaPtr[0] : 1.0f;
  const int rbase = wave * 16 + ((lane < 16) ? 0 : 8);
#pragma unroll
  for (int nt = 0; nt < BN / 16; ++nt) {
    int col = n0 + nt * 16 + (lane & 15);
    float bval = bias ? bias[col] : 0.f;
#pragma unroll
    for (int i = 0; i < 8; ++i) {
      int row = m0 + rbase + i;
      float vv = acc[nt][i] * alpha + bval;
      long idx = (long)bz * cBatch + (long)row * N + col;
      if (Cf) Cf[idx] = vv;
      if (Ch) Ch[idx] = (__bf16)vv;
      if (ChT) {
        int b = row / tS, s = row - b * tS;
        ChT[(long)b * tBatch + (long)col * tS + s] = (__bf16)vv;
      }
    }
  }
}

// ---------------------------------------------------------------------------
__global__ __launch_bounds__(256) void cast_bf16_kernel(
    const float* __restrict__ in, __bf16* __restrict__ out, long n) {
  long i = (long)blockIdx.x * 256 + threadIdx.x;
  if (i < n) out[i] = (__bf16)in[i];
}

// [K,N] fp32 -> [N,K] bf16 (32x32 tiles via LDS)
__global__ __launch_bounds__(256) void cast_transpose_kernel(
    const float* __restrict__ in, __bf16* __restrict__ out, int K, int N) {
  __shared__ float tile[32][33];
  int bx = blockIdx.x * 32;  // N origin
  int by = blockIdx.y * 32;  // K origin
  int tx = threadIdx.x & 31, ty = threadIdx.x >> 5;  // 32 x 8
#pragma unroll
  for (int i = 0; i < 32; i += 8)
    tile[ty + i][tx] = in[(long)(by + ty + i) * N + (bx + tx)];
  __syncthreads();
#pragma unroll
  for (int i = 0; i < 32; i += 8)
    out[(long)(bx + ty + i) * K + (by + tx)] = (__bf16)tile[tx][ty + i];
}

__global__ __launch_bounds__(256) void zero_f32_kernel(float* __restrict__ p, long n) {
  long i = (long)blockIdx.x * 256 + threadIdx.x;
  if (i < n) p[i] = 0.f;
}

// relevancia[row] = sigmoid(max over R columns of punt[row,:])
__global__ __launch_bounds__(256) void rowmax_sigmoid_kernel(
    const float* __restrict__ punt, float* __restrict__ relev, int R) {
  __shared__ float red[256];
  long row = blockIdx.x;
  float m = -3.402823466e38f;
  for (int c = threadIdx.x; c < R; c += 256) m = fmaxf(m, punt[row * (long)R + c]);
  red[threadIdx.x] = m;
  __syncthreads();
  for (int o = 128; o > 0; o >>= 1) {
    if (threadIdx.x < o) red[threadIdx.x] = fmaxf(red[threadIdx.x], red[threadIdx.x + o]);
    __syncthreads();
  }
  if (threadIdx.x == 0) relev[row] = 1.f / (1.f + __expf(-red[0]));
}

// row-wise softmax over n columns; bf16 probabilities out
__global__ __launch_bounds__(256) void softmax_kernel(
    const float* __restrict__ S, __bf16* __restrict__ P, int n) {
  __shared__ float red[256];
  long row = blockIdx.x;
  const float* s = S + row * (long)n;
  float m = -3.402823466e38f;
  for (int c = threadIdx.x; c < n; c += 256) m = fmaxf(m, s[c]);
  red[threadIdx.x] = m;
  __syncthreads();
  for (int o = 128; o > 0; o >>= 1) {
    if (threadIdx.x < o) red[threadIdx.x] = fmaxf(red[threadIdx.x], red[threadIdx.x + o]);
    __syncthreads();
  }
  m = red[0];
  __syncthreads();
  float sum = 0.f;
  for (int c = threadIdx.x; c < n; c += 256) sum += __expf(s[c] - m);
  red[threadIdx.x] = sum;
  __syncthreads();
  for (int o = 128; o > 0; o >>= 1) {
    if (threadIdx.x < o) red[threadIdx.x] += red[threadIdx.x + o];
    __syncthreads();
  }
  float inv = 1.f / red[0];
  __bf16* p = P + row * (long)n;
  for (int c = threadIdx.x; c < n; c += 256) p[c] = (__bf16)(__expf(s[c] - m) * inv);
}

// x = ctx*(1+relev); LayerNorm over D; atomic mean-pool over S into out[b,:]
__global__ __launch_bounds__(256) void ln_pool_kernel(
    const float* __restrict__ ctx, const float* __restrict__ relev,
    const float* __restrict__ gamma, const float* __restrict__ beta,
    float* __restrict__ out, int S, int D) {
  __shared__ float red[256];
  long row = blockIdx.x;  // b*S + s
  int b = (int)(row / S);
  const float* x = ctx + row * (long)D;
  float boost = 1.f + relev[row];

  float sum = 0.f;
  for (int c = threadIdx.x; c < D; c += 256) sum += x[c] * boost;
  red[threadIdx.x] = sum;
  __syncthreads();
  for (int o = 128; o > 0; o >>= 1) {
    if (threadIdx.x < o) red[threadIdx.x] += red[threadIdx.x + o];
    __syncthreads();
  }
  float mu = red[0] / D;
  __syncthreads();

  float vs = 0.f;
  for (int c = threadIdx.x; c < D; c += 256) {
    float d = x[c] * boost - mu;
    vs += d * d;
  }
  red[threadIdx.x] = vs;
  __syncthreads();
  for (int o = 128; o > 0; o >>= 1) {
    if (threadIdx.x < o) red[threadIdx.x] += red[threadIdx.x + o];
    __syncthreads();
  }
  float rstd = rsqrtf(red[0] / D + 1e-5f);
  float invS = 1.f / (float)S;
  for (int c = threadIdx.x; c < D; c += 256) {
    float v = (x[c] * boost - mu) * rstd * gamma[c] + beta[c];
    atomicAdd(&out[(long)b * D + c], v * invS);
  }
}

// ---------------------------------------------------------------------------
extern "C" void kernel_launch(void* const* d_in, const int* in_sizes, int n_in,
                              void* d_out, int out_size, void* d_ws, size_t ws_size,
                              hipStream_t stream) {
  (void)in_sizes; (void)n_in; (void)ws_size;
  const int BB = 16, S = 1024, D = 1024, R = 512, E = 768;
  const int M = BB * S;  // 16384

  const float* x      = (const float*)d_in[0];
  const float* emb    = (const float*)d_in[1];
  const float* Wq     = (const float*)d_in[2];
  const float* bq     = (const float*)d_in[3];
  const float* Wk     = (const float*)d_in[4];
  const float* bk     = (const float*)d_in[5];
  const float* Wv     = (const float*)d_in[6];
  const float* bv     = (const float*)d_in[7];
  const float* Wpr    = (const float*)d_in[8];
  const float* bpr    = (const float*)d_in[9];
  const float* escala = (const float*)d_in[10];
  const float* gamma  = (const float*)d_in[11];
  const float* beta   = (const float*)d_in[12];
  float* out = (float*)d_out;

  char* ws = (char*)d_ws;
  size_t off = 0;
  auto alloc = [&](size_t bytes) {
    size_t o = off;
    off += (bytes + 255) & ~(size_t)255;
    return o;
  };
  __bf16* xb    = (__bf16*)(ws + alloc((size_t)M * D * 2));
  __bf16* wqT   = (__bf16*)(ws + alloc((size_t)D * D * 2));  // [N=D, K=D]
  __bf16* wkT   = (__bf16*)(ws + alloc((size_t)D * D * 2));
  __bf16* wvT   = (__bf16*)(ws + alloc((size_t)D * D * 2));
  __bf16* wprT  = (__bf16*)(ws + alloc((size_t)D * E * 2));  // [N=D, K=E]
  __bf16* embb  = (__bf16*)(ws + alloc((size_t)R * E * 2));
  __bf16* qb    = (__bf16*)(ws + alloc((size_t)M * D * 2));  // row-major
  __bf16* kb    = (__bf16*)(ws + alloc((size_t)M * D * 2));  // row-major
  __bf16* vT    = (__bf16*)(ws + alloc((size_t)M * D * 2));  // per-batch [D,S]
  __bf16* rpb   = (__bf16*)(ws + alloc((size_t)R * D * 2));  // row-major [R,D]
  float*  relev = (float*)(ws + alloc((size_t)M * 4));
  // scores region (64MB) doubles as punt_lex region (32MB): punt is consumed
  // by rowmax_sigmoid before scores are written.
  float*  scores = (float*)(ws + alloc((size_t)BB * S * S * 4));
  float*  punt   = scores;
  __bf16* probs  = (__bf16*)(ws + alloc((size_t)BB * S * S * 2));
  float*  ctx    = (float*)(ws + alloc((size_t)BB * S * D * 4));

  // 0) zero output accumulator
  zero_f32_kernel<<<((long)out_size + 255) / 256, 256, 0, stream>>>(out, out_size);

  // 1) casts (A operands stay row-major; all B operands become [N,K])
  cast_bf16_kernel<<<((long)M * D + 255) / 256, 256, 0, stream>>>(x, xb, (long)M * D);
  cast_bf16_kernel<<<((long)R * E + 255) / 256, 256, 0, stream>>>(emb, embb, (long)R * E);
  {
    dim3 gt(D / 32, D / 32);
    cast_transpose_kernel<<<gt, 256, 0, stream>>>(Wq, wqT, D, D);
    cast_transpose_kernel<<<gt, 256, 0, stream>>>(Wk, wkT, D, D);
    cast_transpose_kernel<<<gt, 256, 0, stream>>>(Wv, wvT, D, D);
    dim3 gp(D / 32, E / 32);
    cast_transpose_kernel<<<gp, 256, 0, stream>>>(Wpr, wprT, E, D);
  }

  // 2) Q/K/V projections: [M,D] = xb @ W^T(stored [D,D]) + bias
  {
    dim3 g(D / BN, M / BM, 1);
    gemm_bf16_kernel<<<g, 256, 0, stream>>>(xb, wqT, bq, nullptr, nullptr, qb, nullptr,
                                            M, D, D, S, 0, 0, 0, 0);
    gemm_bf16_kernel<<<g, 256, 0, stream>>>(xb, wkT, bk, nullptr, nullptr, kb, nullptr,
                                            M, D, D, S, 0, 0, 0, 0);
    // V is written transposed per batch: vT[b][d][s]
    gemm_bf16_kernel<<<g, 256, 0, stream>>>(xb, wvT, bv, nullptr, nullptr, nullptr, vT,
                                            M, D, D, S, 0, 0, 0, (long)S * D);
  }

  // 3) riesgo_proy: [R,D] = emb @ Wpr + bpr -> bf16 row-major
  {
    dim3 g(D / BN, R / BM, 1);
    gemm_bf16_kernel<<<g, 256, 0, stream>>>(embb, wprT, bpr, nullptr, nullptr, rpb, nullptr,
                                            R, D, E, S, 0, 0, 0, 0);
  }

  // 4) punt_lex: [M,R] = q @ rp^T (rp stored [R,D]) -> fp32; rowmax+sigmoid
  {
    dim3 g(R / BN, M / BM, 1);
    gemm_bf16_kernel<<<g, 256, 0, stream>>>(qb, rpb, nullptr, nullptr, punt, nullptr, nullptr,
                                            M, R, D, S, 0, 0, 0, 0);
    rowmax_sigmoid_kernel<<<M, 256, 0, stream>>>(punt, relev, R);
  }

  // 5) attention scores: per batch [S,S] = escala * q_b @ k_b^T (k stored [S,D])
  {
    dim3 g(S / BN, S / BM, BB);
    gemm_bf16_kernel<<<g, 256, 0, stream>>>(qb, kb, nullptr, escala, scores, nullptr, nullptr,
                                            S, S, D, S,
                                            (long)S * D, (long)S * D, (long)S * S, 0);
    softmax_kernel<<<(long)BB * S, 256, 0, stream>>>(scores, probs, S);
  }

  // 6) context: per batch [S,D] = probs_b @ v_b  (v stored transposed [D,S])
  {
    dim3 g(D / BN, S / BM, BB);
    gemm_bf16_kernel<<<g, 256, 0, stream>>>(probs, vT, nullptr, nullptr, ctx, nullptr, nullptr,
                                            S, D, S, S,
                                            (long)S * S, (long)S * D, (long)S * D, 0);
  }

  // 7) boost + LayerNorm + mean-pool over S into out[B,D]
  ln_pool_kernel<<<(long)BB * S, 256, 0, stream>>>(ctx, relev, gamma, beta, out, S, D);
}
